// TreeTransformer_16870631539431
// MI455X (gfx1250) — matrix operand order repaired and checked by reading, not compiled, not run on hardware
//
#include <hip/hip_runtime.h>
#include <hip/hip_bf16.h>

typedef _Float16 f16;
typedef __attribute__((ext_vector_type(16))) _Float16 v16h;
typedef __attribute__((ext_vector_type(8)))  float    v8f;
typedef __attribute__((ext_vector_type(4)))  unsigned int u32x4;
typedef __attribute__((ext_vector_type(8)))  int      i32x8;
typedef __attribute__((ext_vector_type(4)))  int      i32x4;

#define HID    128
#define NVALID 121
#define QKV3   384

__device__ __forceinline__ v8f wmma32(v16h a, v16h b, v8f c) {
  return __builtin_amdgcn_wmma_f32_16x16x32_f16(false, a, false, b, (short)0, c, false, false);
}

// ---- WMMA fragment loaders (layouts per CDNA5 ISA, wave32) ----
// A 16x32 f16: lane = row; regs cover k = {8*half..8*half+7} and {16+8*half..+7}
// => two contiguous 16B loads.
__device__ __forceinline__ v16h a_f16(const f16* row, int half) {
  union { v16h v; uint4 q[2]; } a;
  a.q[0] = *(const uint4*)(row + (half << 3));
  a.q[1] = *(const uint4*)(row + 16 + (half << 3));
  return a.v;
}
__device__ __forceinline__ v16h a_f32(const float* row, int half) {
  const float4 f0 = *(const float4*)(row + (half << 3));
  const float4 f1 = *(const float4*)(row + 4 + (half << 3));
  const float4 f2 = *(const float4*)(row + 16 + (half << 3));
  const float4 f3 = *(const float4*)(row + 20 + (half << 3));
  v16h a;
  a[0]=(f16)f0.x; a[1]=(f16)f0.y; a[2]=(f16)f0.z; a[3]=(f16)f0.w;
  a[4]=(f16)f1.x; a[5]=(f16)f1.y; a[6]=(f16)f1.z; a[7]=(f16)f1.w;
  a[8]=(f16)f2.x; a[9]=(f16)f2.y; a[10]=(f16)f2.z; a[11]=(f16)f2.w;
  a[12]=(f16)f3.x; a[13]=(f16)f3.y; a[14]=(f16)f3.z; a[15]=(f16)f3.w;
  return a;
}
__device__ __forceinline__ v16h a_zero() {
  v16h a;
#pragma unroll
  for (int i = 0; i < 16; ++i) a[i] = (f16)0.f;
  return a;
}
// B 32x16 from n-major source (elem(k,n) = base[n*ldk + k]):
// lane n = ln16, k = {8*half..+7} and {16+8*half..+7} => two 16B loads.
__device__ __forceinline__ v16h b_nmaj_f16(const f16* base, int ldk, int ln16, int half) {
  const f16* p = base + ln16 * ldk + (half << 3);
  union { v16h v; uint4 q[2]; } b;
  b.q[0] = *(const uint4*)(p);
  b.q[1] = *(const uint4*)(p + 16);
  return b.v;
}

// Row-wise LayerNorm over 128 cols held as acc[ntile][r] in C-layout; writes f16.
__device__ __forceinline__ void ln_store(float acc[8][8], const float* w, const float* b,
                                         f16* dst, int m0, int ln16, int half) {
#pragma unroll
  for (int r = 0; r < 8; ++r) {
    float s = 0.f;
#pragma unroll
    for (int nt = 0; nt < 8; ++nt) s += acc[nt][r];
#pragma unroll
    for (int t = 1; t < 16; t <<= 1) s += __shfl_xor(s, t);
    const float mean = s * (1.f / 128.f);
    float vs = 0.f;
#pragma unroll
    for (int nt = 0; nt < 8; ++nt) { const float d = acc[nt][r] - mean; vs += d * d; }
#pragma unroll
    for (int t = 1; t < 16; t <<= 1) vs += __shfl_xor(vs, t);
    const float rstd = rsqrtf(vs * (1.f / 128.f) + 1e-5f);
    const int row = m0 + r + 8 * half;
#pragma unroll
    for (int nt = 0; nt < 8; ++nt) {
      const int col = nt * 16 + ln16;
      dst[row * HID + col] = (f16)((acc[nt][r] - mean) * rstd * w[col] + b[col]);
    }
  }
}

// ---- Tensor Data Mover: 1-D 32KB tile (4096 x 8B) between LDS and global ----
// 6-arg builtin form: (g0, g1, g2, g3, g4, cpol)
__device__ __forceinline__ void tdm_xtile(bool store, unsigned lds_addr, const void* gptr) {
  const unsigned long long ga = (unsigned long long)gptr;
  u32x4 g0;
  g0[0] = 1u;                                        // count=1, user mode
  g0[1] = lds_addr;                                  // LDS byte address
  g0[2] = (unsigned)(ga & 0xFFFFFFFFu);              // global_addr[31:0]
  g0[3] = (unsigned)((ga >> 32) & 0x1FFFFFFu) | (2u << 30); // [56:32] | type=2
  i32x8 g1;
  g1[0] = (3 << 16);              // data_size = 8B, no multicast/pad/iterate
  g1[1] = (int)(4096u << 16);     // tensor_dim0[15:0] = 4096 (in [79:48])
  g1[2] = (1 << 16);              // tensor_dim0 hi = 0 | tensor_dim1 = 1
  g1[3] = (int)(4096u << 16);     // tensor_dim1 hi = 0 | tile_dim0 = 4096
  g1[4] = 0;                      // tile_dim1 = 0 (unused), tile_dim2 = 0
  g1[5] = 4096;                   // tensor_dim0_stride low32
  g1[6] = 0;
  g1[7] = 0;
  i32x4 z; z[0] = 0; z[1] = 0; z[2] = 0; z[3] = 0;
  i32x8 z8; 
#pragma unroll
  for (int i = 0; i < 8; ++i) z8[i] = 0;
  if (store) __builtin_amdgcn_tensor_store_from_lds(g0, g1, z, z, z8, 0);
  else       __builtin_amdgcn_tensor_load_to_lds(g0, g1, z, z, z8, 0);
}

// ---- weight conversion: f32 -> f16, plain and transposed ----
__global__ __launch_bounds__(256) void cvt_plain(const float* __restrict__ s,
                                                 f16* __restrict__ d, int n) {
  const int i = blockIdx.x * 256 + threadIdx.x;
  if (i < n) d[i] = (f16)s[i];
}
__global__ __launch_bounds__(256) void cvt_trans(const float* __restrict__ s,
                                                 f16* __restrict__ d, int K, int N) {
  const int i = blockIdx.x * 256 + threadIdx.x;   // s[K][N] -> d[N][K]
  if (i < K * N) {
    const int k = i / N, n = i - k * N;
    d[n * K + k] = (f16)s[i];
  }
}

// ---- Tree positional encoding from tree 0 (identical structure per tree) ----
__global__ __launch_bounds__(128) void build_pe(const long long* __restrict__ adj,
                                                float* __restrict__ pe) {
  const int i = threadIdx.x;
  int cols[4]; int nc = 0;
  if (i > 0 && i < NVALID) {
    int node = i;
    int d = (node < 4) ? 1 : (node < 13) ? 2 : (node < 40) ? 3 : 4;
    while (node > 0 && nc < 4) {
      const long long* row = adj + (long long)(node - 1) * 3;
      cols[nc++] = 3 * (d - 1) + (int)row[2];
      node = (int)row[0];
      --d;
    }
  }
  for (int c = 0; c < HID; ++c) {
    float v = 0.f;
    for (int j = 0; j < nc; ++j) if (cols[j] == c) v = 1.f;
    pe[i * HID + c] = v;
  }
}

// ---- x = (forest @ w_in + b_in + pe)[perm] -> f16 X, padded to 128 rows ----
__global__ __launch_bounds__(256) void input_proj(
    const float* __restrict__ forest, const long long* __restrict__ perm,
    const f16* __restrict__ wt_in, const float* __restrict__ b_in,
    const float* __restrict__ pe, f16* __restrict__ X) {
  const int seq = blockIdx.x;
  const int tid = threadIdx.x;
  const int lane = tid & 31, wave = tid >> 5;
  const int ln16 = lane & 15, half = lane >> 4;
  const int m0 = wave * 16;

  const int arow = m0 + ln16;
  const bool avalid = arow < NVALID;
  const long long asrc = avalid ? perm[arow] : 0;
  const float* aptr = forest + ((long long)seq * NVALID + asrc) * 64;

  for (int nt = 0; nt < 8; ++nt) {
    v8f c = {};
#pragma unroll
    for (int kt = 0; kt < 2; ++kt) {
      v16h a = avalid ? a_f32(aptr + kt * 32, half) : a_zero();
      v16h b = b_nmaj_f16(wt_in + (nt * 16) * 64 + kt * 32, 64, ln16, half);
      c = wmma32(a, b, c);
    }
    const int col = nt * 16 + ln16;
    const float bias = b_in[col];
#pragma unroll
    for (int r = 0; r < 8; ++r) {
      const int row = m0 + r + 8 * half;
      f16 o = (f16)0.f;
      if (row < NVALID) {
        const long long src = perm[row];
        o = (f16)(c[r] + bias + pe[src * HID + col]);
      }
      X[(long long)seq * HID * HID + row * HID + col] = o;
    }
  }
}

// ---- one encoder layer per sequence; 192KB LDS; TDM in/out ----
__global__ __launch_bounds__(256) void encoder_layer(
    f16* __restrict__ X,
    const f16* __restrict__ wqkv, const float* __restrict__ bqkv,
    const f16* __restrict__ wo,   const float* __restrict__ bo,
    const float* __restrict__ ln1w, const float* __restrict__ ln1b,
    const f16* __restrict__ w1,   const float* __restrict__ b1,
    const f16* __restrict__ w2,   const float* __restrict__ b2,
    const float* __restrict__ ln2w, const float* __restrict__ ln2b) {
  extern __shared__ f16 smem[];
  f16* xs   = smem;                   // [128][128] activations (32KB)
  f16* qk   = xs + 128 * 128;         // [128][256] Q|K rows   (64KB)
  f16* vT   = qk + 128 * 256;         // [128 dims][128 keys]  (32KB)
  f16* ps   = vT + 128 * 128;         // [128][128] P / ffn-h  (32KB)
  f16* ao   = ps + 128 * 128;         // [128][128] attn out   (32KB)

  const int seq = blockIdx.x;
  const int tid = threadIdx.x;
  const int lane = tid & 31, wave = tid >> 5;
  const int ln16 = lane & 15, half = lane >> 4;
  const int m0 = wave * 16;
  f16* Xg = X + (long long)seq * 128 * 128;

  // async DMA the whole x tile into LDS (one wave issues; TENSORcnt drains it)
  if (wave == 0) {
    tdm_xtile(false, (unsigned)(unsigned long long)xs, Xg);
    __builtin_amdgcn_s_wait_tensorcnt(0);
  }
  __syncthreads();

  // qkv = x @ Wqkv^T + b.  Q,K rows -> qk; V packed transposed -> vT.
  for (int nt = 0; nt < 16; ++nt) {       // Q and K columns 0..255
    v8f c = {};
#pragma unroll
    for (int kt = 0; kt < 4; ++kt) {
      v16h a = a_f16(xs + (m0 + ln16) * HID + kt * 32, half);
      v16h b = b_nmaj_f16(wqkv + (nt * 16) * HID + kt * 32, HID, ln16, half);
      c = wmma32(a, b, c);
    }
    const int col = nt * 16 + ln16;
    const float bias = bqkv[col];
#pragma unroll
    for (int r = 0; r < 8; ++r)
      qk[(m0 + r + 8 * half) * 256 + col] = (f16)(c[r] + bias);
  }
  for (int nt = 16; nt < 24; ++nt) {      // V columns 256..383, store transposed
    v8f c = {};
#pragma unroll
    for (int kt = 0; kt < 4; ++kt) {
      v16h a = a_f16(xs + (m0 + ln16) * HID + kt * 32, half);
      v16h b = b_nmaj_f16(wqkv + (nt * 16) * HID + kt * 32, HID, ln16, half);
      c = wmma32(a, b, c);
    }
    const int col = nt * 16 + ln16;       // 256..383
    const float bias = bqkv[col];
    union { uint4 q; f16 h[8]; } pk;
#pragma unroll
    for (int r = 0; r < 8; ++r) pk.h[r] = (f16)(c[r] + bias);
    // rows m0+8*half .. +7 are contiguous -> single 16B store
    *(uint4*)(vT + (col - 256) * 128 + m0 + (half << 3)) = pk.q;
  }
  __syncthreads();

  const float scale = 0.1767766953f;  // 1/sqrt(32)
  for (int h = 0; h < 4; ++h) {
    v8f sc[8];
#pragma unroll
    for (int nt = 0; nt < 8; ++nt) {
      v16h a = a_f16(qk + (m0 + ln16) * 256 + h * 32, half);                       // Q
      v16h b = b_nmaj_f16(qk + (nt * 16) * 256 + 128 + h * 32, 256, ln16, half);   // K^T
      v8f z = {};
      sc[nt] = wmma32(a, b, z);
    }
#pragma unroll
    for (int r = 0; r < 8; ++r) {          // masked row softmax
      float mx = -3.0e38f;
#pragma unroll
      for (int nt = 0; nt < 8; ++nt) {
        const int col = nt * 16 + ln16;
        const float v = (col < NVALID) ? sc[nt][r] * scale : -3.0e38f;
        sc[nt][r] = v;
        mx = fmaxf(mx, v);
      }
#pragma unroll
      for (int t = 1; t < 16; t <<= 1) mx = fmaxf(mx, __shfl_xor(mx, t));
      float sum = 0.f;
#pragma unroll
      for (int nt = 0; nt < 8; ++nt) { const float e = __expf(sc[nt][r] - mx); sc[nt][r] = e; sum += e; }
#pragma unroll
      for (int t = 1; t < 16; t <<= 1) sum += __shfl_xor(sum, t);
      const float inv = 1.f / sum;
      const int row = m0 + r + 8 * half;
#pragma unroll
      for (int nt = 0; nt < 8; ++nt)
        ps[row * 128 + nt * 16 + ln16] = (f16)(sc[nt][r] * inv);
    }
    // O = P @ V  (wave-local rows; vT gives contiguous B fragments)
#pragma unroll
    for (int nd = 0; nd < 2; ++nd) {
      v8f c = {};
#pragma unroll
      for (int kt = 0; kt < 4; ++kt) {
        v16h a = a_f16(ps + (m0 + ln16) * 128 + kt * 32, half);
        v16h b = b_nmaj_f16(vT + (h * 32 + nd * 16) * 128 + kt * 32, 128, ln16, half);
        c = wmma32(a, b, c);
      }
      const int col = h * 32 + nd * 16 + ln16;
#pragma unroll
      for (int r = 0; r < 8; ++r)
        ao[(m0 + r + 8 * half) * 128 + col] = (f16)c[r];
    }
  }

  // out-proj + residual + LN1 (wave-local rows)
  float acc[8][8];
#pragma unroll
  for (int nt = 0; nt < 8; ++nt) {
    v8f c = {};
#pragma unroll
    for (int kt = 0; kt < 4; ++kt) {
      v16h a = a_f16(ao + (m0 + ln16) * 128 + kt * 32, half);
      v16h b = b_nmaj_f16(wo + (nt * 16) * HID + kt * 32, HID, ln16, half);
      c = wmma32(a, b, c);
    }
    const int col = nt * 16 + ln16;
    const float bias = bo[col];
#pragma unroll
    for (int r = 0; r < 8; ++r) {
      const int row = m0 + r + 8 * half;
      acc[nt][r] = c[r] + bias + (float)xs[row * HID + col];
    }
  }
  ln_store(acc, ln1w, ln1b, xs, m0, ln16, half);

  // FFN: h1 = relu(x @ w1^T + b1) -> ps
#pragma unroll
  for (int nt = 0; nt < 8; ++nt) {
    v8f c = {};
#pragma unroll
    for (int kt = 0; kt < 4; ++kt) {
      v16h a = a_f16(xs + (m0 + ln16) * HID + kt * 32, half);
      v16h b = b_nmaj_f16(w1 + (nt * 16) * HID + kt * 32, HID, ln16, half);
      c = wmma32(a, b, c);
    }
    const int col = nt * 16 + ln16;
    const float bias = b1[col];
#pragma unroll
    for (int r = 0; r < 8; ++r)
      ps[(m0 + r + 8 * half) * 128 + col] = (f16)fmaxf(c[r] + bias, 0.f);
  }
  // ff2 + residual + LN2
#pragma unroll
  for (int nt = 0; nt < 8; ++nt) {
    v8f c = {};
#pragma unroll
    for (int kt = 0; kt < 4; ++kt) {
      v16h a = a_f16(ps + (m0 + ln16) * 128 + kt * 32, half);
      v16h b = b_nmaj_f16(w2 + (nt * 16) * HID + kt * 32, HID, ln16, half);
      c = wmma32(a, b, c);
    }
    const int col = nt * 16 + ln16;
    const float bias = b2[col];
#pragma unroll
    for (int r = 0; r < 8; ++r) {
      const int row = m0 + r + 8 * half;
      acc[nt][r] = c[r] + bias + (float)xs[row * HID + col];
    }
  }
  ln_store(acc, ln2w, ln2b, xs, m0, ln16, half);
  __syncthreads();
  // async DMA the tile back to global
  if (wave == 0) {
    tdm_xtile(true, (unsigned)(unsigned long long)xs, Xg);
    __builtin_amdgcn_s_wait_tensorcnt(0);
  }
}

// ---- out = LN( xflat @ w_out + b_out ): 64 blocks x 16 rows, wave = N-tile ----
__global__ __launch_bounds__(256) void out_head(
    const f16* __restrict__ X, const f16* __restrict__ wt_out,
    const float* __restrict__ b_out, const float* __restrict__ nw,
    const float* __restrict__ nb, float* __restrict__ out) {
  __shared__ float cs[16][132];
  const int tid = threadIdx.x;
  const int lane = tid & 31, wave = tid >> 5;
  const int ln16 = lane & 15, half = lane >> 4;
  const int m0 = blockIdx.x * 16;

  const f16* arow = X + (long long)(m0 + ln16) * 16384;    // k<15488 => valid rows only
  const f16* brow = wt_out + (long long)(wave * 16) * 15488;
  v8f c = {};
#pragma unroll 4
  for (int kt = 0; kt < 484; ++kt) {
    v16h a = a_f16(arow + kt * 32, half);
    v16h b = b_nmaj_f16(brow + kt * 32, 15488, ln16, half);
    c = wmma32(a, b, c);
  }
  const int col = wave * 16 + ln16;
  const float bias = b_out[col];
#pragma unroll
  for (int r = 0; r < 8; ++r) cs[r + 8 * half][col] = c[r] + bias;
  __syncthreads();

  const int row = tid >> 4;
  const int l = tid & 15;
  float v[8]; float s = 0.f;
#pragma unroll
  for (int j = 0; j < 8; ++j) { v[j] = cs[row][l + 16 * j]; s += v[j]; }
#pragma unroll
  for (int t = 1; t < 16; t <<= 1) s += __shfl_xor(s, t);
  const float mean = s * (1.f / 128.f);
  float vs = 0.f;
#pragma unroll
  for (int j = 0; j < 8; ++j) { const float d = v[j] - mean; vs += d * d; }
#pragma unroll
  for (int t = 1; t < 16; t <<= 1) vs += __shfl_xor(vs, t);
  const float rstd = rsqrtf(vs * (1.f / 128.f) + 1e-5f);
#pragma unroll
  for (int j = 0; j < 8; ++j) {
    const int cc = l + 16 * j;
    out[(long long)(m0 + row) * 128 + cc] = (v[j] - mean) * rstd * nw[cc] + nb[cc];
  }
}

extern "C" void kernel_launch(void* const* d_in, const int* in_sizes, int n_in,
                              void* d_out, int out_size, void* d_ws, size_t ws_size,
                              hipStream_t stream) {
  (void)in_sizes; (void)n_in; (void)out_size; (void)ws_size;
  const float*     forest = (const float*)d_in[0];
  const long long* adj    = (const long long*)d_in[1];
  const long long* perm   = (const long long*)d_in[2];
  const float* w_in  = (const float*)d_in[3];
  const float* b_in  = (const float*)d_in[4];
  const float* wqkv  = (const float*)d_in[5];
  const float* bqkv  = (const float*)d_in[6];
  const float* wo    = (const float*)d_in[7];
  const float* bo    = (const float*)d_in[8];
  const float* l1w   = (const float*)d_in[9];
  const float* l1b   = (const float*)d_in[10];
  const float* w1    = (const float*)d_in[11];
  const float* b1    = (const float*)d_in[12];
  const float* w2    = (const float*)d_in[13];
  const float* b2    = (const float*)d_in[14];
  const float* l2w   = (const float*)d_in[15];
  const float* l2b   = (const float*)d_in[16];
  const float* w_out = (const float*)d_in[17];
  const float* b_out = (const float*)d_in[18];
  const float* nw    = (const float*)d_in[19];
  const float* nb    = (const float*)d_in[20];

  char* wsb = (char*)d_ws;
  float* pe = (float*)wsb;                              // 64KB
  f16* X    = (f16*)(wsb + 65536);                      // 1024*128*128 f16 = 32MB
  f16* wt   = (f16*)(wsb + 65536 + (size_t)33554432);   // f16 weight mirror
  f16* wt_in  = wt;                                     // [128][64]  (transposed)
  f16* wt_qkv = wt_in + 128 * 64;                       // [2][384][128]
  f16* wt_o   = wt_qkv + 2 * QKV3 * HID;                // [2][128][128]
  f16* wt_1   = wt_o + 2 * HID * HID;
  f16* wt_2   = wt_1 + 2 * HID * HID;
  f16* wt_out = wt_2 + 2 * HID * HID;                   // [128][15488] (transposed)

  // one-time weight conversion/transposition to f16 (L2-resident afterwards)
  cvt_trans<<<(64 * 128 + 255) / 256, 256, 0, stream>>>(w_in, wt_in, 64, 128);
  cvt_plain<<<(2 * QKV3 * HID + 255) / 256, 256, 0, stream>>>(wqkv, wt_qkv, 2 * QKV3 * HID);
  cvt_plain<<<(2 * HID * HID + 255) / 256, 256, 0, stream>>>(wo, wt_o, 2 * HID * HID);
  cvt_plain<<<(2 * HID * HID + 255) / 256, 256, 0, stream>>>(w1, wt_1, 2 * HID * HID);
  cvt_plain<<<(2 * HID * HID + 255) / 256, 256, 0, stream>>>(w2, wt_2, 2 * HID * HID);
  cvt_trans<<<(15488 * 128 + 255) / 256, 256, 0, stream>>>(w_out, wt_out, 15488, 128);

  build_pe<<<1, 128, 0, stream>>>(adj, pe);
  input_proj<<<1024, 256, 0, stream>>>(forest, perm, wt_in, b_in, pe, X);

  const size_t lds_bytes = (size_t)(128 * 128 + 128 * 256 + 3 * 128 * 128) * sizeof(f16); // 192KB
  for (int l = 0; l < 2; ++l) {
    encoder_layer<<<1024, 256, lds_bytes, stream>>>(
        X,
        wt_qkv + (size_t)l * QKV3 * HID, bqkv + (size_t)l * QKV3,
        wt_o   + (size_t)l * HID * HID,  bo   + (size_t)l * HID,
        l1w + (size_t)l * HID, l1b + (size_t)l * HID,
        wt_1   + (size_t)l * HID * HID,  b1   + (size_t)l * HID,
        wt_2   + (size_t)l * HID * HID,  b2   + (size_t)l * HID,
        l2w + (size_t)l * HID, l2b + (size_t)l * HID);
  }
  out_head<<<64, 256, 0, stream>>>(X, wt_out, b_out, nw, nb, (float*)d_out);
}
